// UNetV2_67997922230615
// MI455X (gfx1250) — matrix operand order repaired and checked by reading, not compiled, not run on hardware
//
#include <hip/hip_runtime.h>

#define NV 500000
#define NTILES (NV / 16)   // 31250, exact
#define WPB 8              // waves per block (256 threads)
#define TPW 8              // tiles per wave  (one LDS weight fill serves 64 tiles)

typedef __attribute__((ext_vector_type(16))) __bf16       v16bf;
typedef __attribute__((ext_vector_type(8)))  float        v8f;
typedef __attribute__((ext_vector_type(4)))  unsigned int v4u;

union Frag { v16bf v; v4u u[2]; };

static __device__ __forceinline__ unsigned short f2bf(float x) {
    union { float f; unsigned u; } v; v.f = x;
    unsigned r = v.u + 0x7FFFu + ((v.u >> 16) & 1u);   // round-to-nearest-even
    return (unsigned short)(r >> 16);
}
static __device__ __forceinline__ float bf2f(unsigned hs) {
    union { unsigned u; float f; } v; v.u = hs << 16; return v.f;
}

// ---------------- prep: fold BN into y = x*alpha + beta ----------------
__global__ void fold_bn(const float* __restrict__ b0, const float* __restrict__ b1,
                        const float* __restrict__ b2, const float* __restrict__ b3,
                        float* __restrict__ bnf) {
    int i = threadIdx.x;            // 128 threads: 4 stages x 32 channels
    int s = i >> 5, c = i & 31;
    const float* p = (s == 0) ? b0 : (s == 1) ? b1 : (s == 2) ? b2 : b3;
    float al = p[0 * 32 + c] * rsqrtf(p[3 * 32 + c] + 1e-3f);
    float be = p[1 * 32 + c] - p[2 * 32 + c] * al;
    bnf[s * 64 + c]      = al;
    bnf[s * 64 + 32 + c] = be;
}

// ---------------- prep: swizzle weights into WMMA B-fragment layout ----------------
// dst linear index = (((k*KT + t)*2 + h)*32 + lane)*8 + v  (dwords, bf16x2 packed)
// lane 0..15 -> K = t*32 + (v>>2)*16 + 2(v&3)(+1), col = h*16 + lane ; lane 16..31 -> K += 8
__global__ void swizzle_w(const float* __restrict__ W, unsigned int* __restrict__ dst, int KT) {
    int i = blockIdx.x * blockDim.x + threadIdx.x;
    int total = 27 * KT * 512;
    if (i >= total) return;
    int v    = i & 7;
    int lane = (i >> 3) & 31;
    int h    = (i >> 8) & 1;
    int kt   = i >> 9;
    int t = kt % KT, k = kt / KT;
    int cin = 32 * KT;
    int kk  = t * 32 + (v >> 2) * 16 + ((lane >> 4) & 1) * 8 + 2 * (v & 3);
    int col = h * 16 + (lane & 15);
    unsigned lo = f2bf(W[(k * cin + kk) * 32 + col]);
    unsigned hi = f2bf(W[(k * cin + kk + 1) * 32 + col]);
    dst[i] = (hi << 16) | lo;
}

// ---------------- prep: f32 -> bf16 features, zero pad rows ----------------
__global__ void cvt_feats(const float* __restrict__ lat, const float* __restrict__ bot,
                          unsigned short* __restrict__ latB, unsigned short* __restrict__ catB,
                          unsigned short* __restrict__ t1B, unsigned short* __restrict__ xB) {
    long i = (long)blockIdx.x * 256 + threadIdx.x;   // over (NV+1)*32
    if (i >= (long)(NV + 1) * 32) return;
    int row = (int)(i >> 5), c = (int)(i & 31);
    if (row < NV) {
        latB[i] = f2bf(lat[i]);
        catB[(size_t)row * 64 + c] = f2bf(bot[i]);   // cat low half = bottom
    } else {                                          // pad row must read as zero
        latB[i] = 0;
        catB[(size_t)row * 64 + c] = 0;
        catB[(size_t)row * 64 + 32 + c] = 0;
        t1B[i] = 0;
        xB[i]  = 0;
    }
}

// ---------------- main: submanifold conv, 1 wave32 = 16 voxels x 32 out channels ----------------
// MODE 0: relu(bn)            -> bf16   (stage1)
// MODE 1: relu(bn + resid)    -> bf16   (stage2, residual = lateral f32)
// MODE 2: relu(bn) + x_red    -> bf16   (stage3, x_red from cat buffer)
// MODE 3: relu(bn)            -> f32    (stage4, final output)
template<int KT, int MODE>
__global__ __launch_bounds__(256)
void subm_conv(const unsigned short* __restrict__ feats,  // (NV+1) rows x 32*KT bf16
               const int* __restrict__ nbr,               // [NV][27], pad = NV
               const unsigned int* __restrict__ wsw,      // swizzled weights (this stage)
               const float* __restrict__ bnf,             // alpha[0..31], beta[32..63]
               const float* __restrict__ resid,
               const unsigned short* __restrict__ catB,
               unsigned short* __restrict__ outB, int ostride, int ooff,
               float* __restrict__ outF) {
    extern __shared__ unsigned int wlds[];
    const int NWD = 27 * KT * 512;                        // dwords of weights
    {   // cooperative weight preload (b128 copies)
        const v4u* s4 = (const v4u*)wsw;
        v4u* d4 = (v4u*)wlds;
        for (int i = threadIdx.x; i < NWD / 4; i += WPB * 32) d4[i] = s4[i];
    }
    __syncthreads();

    const int wave = threadIdx.x >> 5;
    const int lane = threadIdx.x & 31;
    const int lh   = (lane >> 4) & 1;                     // lane half
    const int cn   = lane & 15;
    const int tb   = blockIdx.x * (WPB * TPW) + wave;     // this wave's first tile

    // per-lane BN constants (channel does not depend on tile)
    const float al0 = bnf[cn],      be0 = bnf[32 + cn];
    const float al1 = bnf[16 + cn], be1 = bnf[48 + cn];

    const char* fbase = (const char*)feats;               // 32-bit voffset addressing
    const int   SH    = (KT == 1) ? 6 : 7;                // row bytes = 64 / 128
    const unsigned lh16 = (unsigned)lh * 16u;

    for (int it = 0; it < TPW; ++it) {
        const int tile = tb + it * WPB;
        if (tile >= NTILES) break;                        // wave-uniform

        const int m0 = tile * 16 + cn;                    // A-matrix row this lane gathers

        v8f acc0 = {}; v8f acc1 = {};
        int idx = nbr[m0 * 27];
        for (int k = 0; k < 27; ++k) {
            int nidx = (k < 26) ? nbr[m0 * 27 + k + 1] : 0;   // prefetch next tap's index
            const unsigned ga = ((unsigned)idx << SH) + lh16; // byte offset of this lane's chunk
#pragma unroll
            for (int t = 0; t < KT; ++t) {
                Frag a, b0, b1;
                // A 16x32 bf16: lanes0-15 hold K 0-7 & 16-23, lanes16-31 hold K 8-15 & 24-31
                a.u[0] = *(const v4u*)(fbase + ga + t * 64);
                a.u[1] = *(const v4u*)(fbase + ga + t * 64 + 32);
                const v4u* wp = (const v4u*)(wlds + (k * KT + t) * 512 + lane * 8);
                b0.u[0] = wp[0];  b0.u[1] = wp[1];            // N = 0..15
                b1.u[0] = wp[64]; b1.u[1] = wp[65];           // N = 16..31 (+256 dwords)
                acc0 = __builtin_amdgcn_wmma_f32_16x16x32_bf16(false, a.v, false, b0.v,
                                                               (short)0, acc0, false, false);
                acc1 = __builtin_amdgcn_wmma_f32_16x16x32_bf16(false, a.v, false, b1.v,
                                                               (short)0, acc1, false, false);
            }
            idx = nidx;
        }

        // D layout: VGPR r, lanes0-15 -> M=r, lanes16-31 -> M=r+8 ; N = lane&15
#pragma unroll
        for (int h = 0; h < 2; ++h) {
            v8f av = h ? acc1 : acc0;
            int c = h * 16 + cn;
            float al = h ? al1 : al0, be = h ? be1 : be0;
#pragma unroll
            for (int r = 0; r < 8; ++r) {
                int m = tile * 16 + r + lh * 8;
                float y = av[r] * al + be;
                if (MODE == 0) {
                    outB[(size_t)m * ostride + ooff + c] = f2bf(fmaxf(y, 0.f));
                } else if (MODE == 1) {
                    float o = fmaxf(y + resid[(size_t)m * 32 + c], 0.f);
                    outB[(size_t)m * ostride + ooff + c] = f2bf(o);
                } else if (MODE == 2) {
                    unsigned pr = *(const unsigned*)(catB + (size_t)m * 64 + 2 * c);
                    float xr = bf2f(pr & 0xFFFFu) + bf2f(pr >> 16);
                    outB[(size_t)m * ostride + ooff + c] = f2bf(fmaxf(y, 0.f) + xr);
                } else {
                    outF[(size_t)m * 32 + c] = fmaxf(y, 0.f);
                }
            }
        }
    }
}

extern "C" void kernel_launch(void* const* d_in, const int* in_sizes, int n_in,
                              void* d_out, int out_size, void* d_ws, size_t ws_size,
                              hipStream_t stream) {
    const float* lat  = (const float*)d_in[0];
    const float* bot  = (const float*)d_in[1];
    const float* Wt1  = (const float*)d_in[2];
    const float* bnt1 = (const float*)d_in[3];
    const float* Wt2  = (const float*)d_in[4];
    const float* bnt2 = (const float*)d_in[5];
    const float* Wm   = (const float*)d_in[6];
    const float* bnm  = (const float*)d_in[7];
    const float* Wi   = (const float*)d_in[8];
    const float* bni  = (const float*)d_in[9];
    const int*   nbr  = (const int*)d_in[10];
    float* out = (float*)d_out;

    char* ws = (char*)d_ws;
    size_t off = 0;
    auto take = [&](size_t b) -> char* {
        char* p = ws + off; off = (off + b + 255) & ~(size_t)255; return p;
    };
    unsigned short* latB = (unsigned short*)take((size_t)(NV + 1) * 32 * 2);
    unsigned short* t1B  = (unsigned short*)take((size_t)(NV + 1) * 32 * 2);
    unsigned short* catB = (unsigned short*)take((size_t)(NV + 1) * 64 * 2);
    unsigned short* xB   = (unsigned short*)take((size_t)(NV + 1) * 32 * 2);
    unsigned int*   wsw  = (unsigned int*)take((size_t)69120 * 4);
    float*          bnf  = (float*)take(4 * 64 * 4);

    // ---- prep ----
    fold_bn<<<1, 128, 0, stream>>>(bnt1, bnt2, bnm, bni, bnf);
    swizzle_w<<<(13824 + 255) / 256, 256, 0, stream>>>(Wt1, wsw,          1);
    swizzle_w<<<(13824 + 255) / 256, 256, 0, stream>>>(Wt2, wsw + 13824,  1);
    swizzle_w<<<(27648 + 255) / 256, 256, 0, stream>>>(Wm,  wsw + 27648,  2);
    swizzle_w<<<(13824 + 255) / 256, 256, 0, stream>>>(Wi,  wsw + 55296,  1);
    long tot = (long)(NV + 1) * 32;
    cvt_feats<<<(unsigned)((tot + 255) / 256), 256, 0, stream>>>(lat, bot, latB, catB, t1B, xB);

    // allow 110 KB dynamic LDS for the 64-channel stage (WGP has 320 KB)
    (void)hipFuncSetAttribute((const void*)subm_conv<2, 2>,
                              hipFuncAttributeMaxDynamicSharedMemorySize, 27 * 2 * 512 * 4);

    const int tilesPerBlock = WPB * TPW;                          // 64
    const int blocks = (NTILES + tilesPerBlock - 1) / tilesPerBlock;  // 489
    const unsigned lds1 = 27 * 512 * 4;       // 55296 B
    const unsigned lds2 = 27 * 2 * 512 * 4;   // 110592 B

    // stage1: relu(bn1(conv(lat)))                       -> t1B (bf16)
    subm_conv<1, 0><<<blocks, 256, lds1, stream>>>(latB, nbr, wsw,         bnf + 0,   nullptr, nullptr,
                                                   t1B, 32, 0, nullptr);
    // stage2: relu(bn2(conv(t1)) + lat) = x_trans        -> catB cols 32..63
    subm_conv<1, 1><<<blocks, 256, lds1, stream>>>(t1B,  nbr, wsw + 13824, bnf + 64,  lat,     nullptr,
                                                   catB, 64, 32, nullptr);
    // stage3: relu(bn_m(conv64(cat))) + x_red(cat)       -> xB (bf16)
    subm_conv<2, 2><<<blocks, 256, lds2, stream>>>(catB, nbr, wsw + 27648, bnf + 128, nullptr, catB,
                                                   xB, 32, 0, nullptr);
    // stage4: relu(bn_i(conv(x)))                        -> d_out (f32)
    subm_conv<1, 3><<<blocks, 256, lds1, stream>>>(xB,   nbr, wsw + 55296, bnf + 192, nullptr, nullptr,
                                                   nullptr, 32, 0, out);
}
